// MY_NonLocalBlockND_61443802137018
// MI455X (gfx1250) — compile-verified, matrix operand order checked by reading, not run on hardware
//
#include <hip/hip_runtime.h>
#include <hip/hip_fp16.h>

typedef __attribute__((ext_vector_type(16))) _Float16 v16h;
typedef __attribute__((ext_vector_type(8)))  float    v8f;

#define NB   8
#define CIN  256
#define CIH  128
#define HH   64
#define WW_  64
#define NN   4096   // H*W
#define MM   1024   // (H/2)*(W/2)
#define EPSB 1e-5f

// ---------------------------------------------------------------------------
// 2x2 average pool on q and v:  [B][C][64][64] -> [B][C][32][32]
// (1x1 conv commutes with avgpool, so pooling first saves 4x conv FLOPs)
// ---------------------------------------------------------------------------
__global__ void pool2_kernel(const float* __restrict__ q,
                             const float* __restrict__ v,
                             float* __restrict__ qp,
                             float* __restrict__ vp) {
  int idx = blockIdx.x * blockDim.x + threadIdx.x;   // < B*C*M
  int m  = idx & (MM - 1);
  int bc = idx >> 10;
  int mh = m >> 5, mw = m & 31;
  size_t base = (size_t)bc * (HH * WW_) + (size_t)(2 * mh) * WW_ + 2 * mw;
  qp[idx] = 0.25f * (q[base] + q[base + 1] + q[base + WW_] + q[base + WW_ + 1]);
  vp[idx] = 0.25f * (v[base] + v[base + 1] + v[base + WW_] + v[base + WW_ + 1]);
}

// ---------------------------------------------------------------------------
// One-shot f32 -> fp16 conversion (used to pre-convert all weight matrices)
// ---------------------------------------------------------------------------
__global__ void f32_to_f16_kernel(const float* __restrict__ in,
                                  _Float16* __restrict__ o, int n) {
  int i = blockIdx.x * blockDim.x + threadIdx.x;
  if (i < n) o[i] = (_Float16)in[i];
}

// ---------------------------------------------------------------------------
// BN constants: scale = gamma*rsqrt(var+eps); shift = (bw-mean)*scale + beta
// ---------------------------------------------------------------------------
__global__ void bn_prep_kernel(const float* __restrict__ gamma,
                               const float* __restrict__ beta,
                               const float* __restrict__ mean,
                               const float* __restrict__ var,
                               const float* __restrict__ bw,
                               float* __restrict__ sc, float* __restrict__ sh) {
  int i = threadIdx.x;   // one block of 256
  float s = gamma[i] * rsqrtf(var[i] + EPSB);
  sc[i] = s;
  sh[i] = (bw[i] - mean[i]) * s + beta[i];
}

// ---------------------------------------------------------------------------
// 1x1 conv as WMMA GEMM: outT[ci][s] = sum_c W[ci][c] * x[b][c][s] + bias[ci]
// Block = 256 threads = 8 waves; wave w owns ci-rows [16w,16w+16).
// A (fp16 weights) fragments: direct global b128 loads.
// B (activations) staged through an LDS fp16 transpose tile [s][c].
// OUT_LAYOUT 0 -> out[b][s][ci] ; 1 -> out[b][ci][s]
// ---------------------------------------------------------------------------
template <int OUT_LAYOUT>
__global__ __launch_bounds__(256)
void conv1x1_f16_kernel(const float* __restrict__ x,       // [B][CIN][S]
                        const _Float16* __restrict__ Wh,   // [CIH][CIN] fp16
                        const float* __restrict__ bias,    // [CIH]
                        _Float16* __restrict__ out, int S) {
  const int b     = blockIdx.y;
  const int s0    = blockIdx.x * 16;
  const int tid   = threadIdx.x;
  const int wave  = tid >> 5;
  const int lane  = tid & 31;
  const int lhalf = lane & 15;
  const int hi    = lane >> 4;

  __shared__ __align__(16) _Float16 xs[16 * 48];  // [s][c], 16B-aligned rows

  const float* xb = x + (size_t)b * CIN * S;

  v8f acc = {};
  for (int c0 = 0; c0 < CIN; c0 += 32) {
    __syncthreads();
    {
      int cc = tid >> 4;      // 0..15
      int ss = tid & 15;
      float f0 = xb[(size_t)(c0 + cc) * S + s0 + ss];
      float f1 = xb[(size_t)(c0 + cc + 16) * S + s0 + ss];
      xs[ss * 48 + cc]      = (_Float16)f0;
      xs[ss * 48 + cc + 16] = (_Float16)f1;
    }
    __syncthreads();

    // A fragment: row ci = 16*wave + lhalf ; K = c0 + hi*8 + {0..7, 16..23}
    const _Float16* wrow = Wh + (size_t)(16 * wave + lhalf) * CIN + c0 + hi * 8;
    v16h a;
    ((uint4*)&a)[0] = *(const uint4*)(wrow);
    ((uint4*)&a)[1] = *(const uint4*)(wrow + 16);

    // B fragment from LDS: col s = lhalf ; K = hi*8 + {0..7, 16..23}
    const _Float16* brow = &xs[lhalf * 48 + hi * 8];
    v16h bf;
    ((uint4*)&bf)[0] = *(const uint4*)(brow);
    ((uint4*)&bf)[1] = *(const uint4*)(brow + 16);

    acc = __builtin_amdgcn_wmma_f32_16x16x32_f16(false, a, false, bf,
                                                 (short)0, acc, false, false);
  }

  const int ciBase = 16 * wave;
#pragma unroll
  for (int j = 0; j < 8; ++j) {
    int ci = ciBase + j + 8 * hi;          // D row
    int s  = s0 + lhalf;                   // D col
    _Float16 hval = (_Float16)(acc[j] + bias[ci]);
    if (OUT_LAYOUT == 0)
      out[((size_t)b * S + s) * CIH + ci] = hval;
    else
      out[((size_t)b * CIH + ci) * S + s] = hval;
  }
}

// ---------------------------------------------------------------------------
// Flash-attention core: per wave, 16 rows of n. Online softmax over M=1024,
// processed in m-chunks of 64 (4 f-tiles) to amortize softmax bookkeeping.
//   f  = theta @ phi^T   (K = CI = 128)
//   y += softmax(f) @ g  (K = m, two K=32 sub-chunks per iteration)
// theta [B][N][CI], phi [B][M][CI], g [B][CI][M] fp16 -> y [B][N][CI] fp16
// Block = 128 threads = 4 independent waves.
// ---------------------------------------------------------------------------
__global__ __launch_bounds__(128)
void attn_kernel(const _Float16* __restrict__ theta,
                 const _Float16* __restrict__ phi,
                 const _Float16* __restrict__ g,
                 _Float16* __restrict__ yout) {
  const int b     = blockIdx.y;
  const int tid   = threadIdx.x;
  const int wave  = tid >> 5;
  const int lane  = tid & 31;
  const int lhalf = lane & 15;
  const int hi    = lane >> 4;
  const int n0    = (blockIdx.x * 4 + wave) * 16;

  // per-wave P tile [16 rows][64 m], row stride 72 halfs (144B, 16B-aligned)
  __shared__ __align__(16) _Float16 pbuf[4][16 * 72];

  // Cache theta A fragments (16 rows x K=128 -> 4 chunks of K=32)
  v16h ta[4];
  const _Float16* trow = theta + ((size_t)b * NN + n0 + lhalf) * CIH;
#pragma unroll
  for (int c = 0; c < 4; ++c) {
    const _Float16* p = trow + 32 * c + hi * 8;
    ((uint4*)&ta[c])[0] = *(const uint4*)(p);
    ((uint4*)&ta[c])[1] = *(const uint4*)(p + 16);
  }

  float mrun[8], lrun[8];
#pragma unroll
  for (int j = 0; j < 8; ++j) { mrun[j] = -1e30f; lrun[j] = 0.0f; }
  v8f yacc[8];
#pragma unroll
  for (int t = 0; t < 8; ++t) yacc[t] = (v8f){};

  _Float16* pb = pbuf[wave];

  for (int m0 = 0; m0 < MM; m0 += 64) {
    // ---- f tiles t=0..3: columns [m0+16t, m0+16t+16)
    v8f f[4];
#pragma unroll
    for (int t = 0; t < 4; ++t) f[t] = (v8f){};
#pragma unroll
    for (int t = 0; t < 4; ++t)
#pragma unroll
      for (int c = 0; c < 4; ++c) {
        const _Float16* pr =
            phi + ((size_t)b * MM + m0 + 16 * t + lhalf) * CIH + 32 * c + hi * 8;
        v16h bf;
        ((uint4*)&bf)[0] = *(const uint4*)(pr);
        ((uint4*)&bf)[1] = *(const uint4*)(pr + 16);
        f[t] = __builtin_amdgcn_wmma_f32_16x16x32_f16(false, ta[c], false, bf,
                                                      (short)0, f[t], false, false);
      }

    // ---- online softmax: row stats per lane-half (row = j + 8*hi)
    float mc[8];
#pragma unroll
    for (int j = 0; j < 8; ++j)
      mc[j] = fmaxf(fmaxf(f[0][j], f[1][j]), fmaxf(f[2][j], f[3][j]));
#pragma unroll
    for (int j = 0; j < 8; ++j)
      for (int d = 1; d < 16; d <<= 1)
        mc[j] = fmaxf(mc[j], __shfl_xor(mc[j], d, 32));

    float scale[8];
#pragma unroll
    for (int j = 0; j < 8; ++j) {
      float mn = fmaxf(mrun[j], mc[j]);
      scale[j] = __expf(mrun[j] - mn);
      mrun[j]  = mn;
    }

    float ls[8];
#pragma unroll
    for (int j = 0; j < 8; ++j) ls[j] = 0.0f;
#pragma unroll
    for (int t = 0; t < 4; ++t)
#pragma unroll
      for (int j = 0; j < 8; ++j) {
        float p = __expf(f[t][j] - mrun[j]);
        f[t][j] = p;
        ls[j] += p;
      }
#pragma unroll
    for (int j = 0; j < 8; ++j)
      for (int d = 1; d < 16; d <<= 1)
        ls[j] += __shfl_xor(ls[j], d, 32);
#pragma unroll
    for (int j = 0; j < 8; ++j) lrun[j] = lrun[j] * scale[j] + ls[j];

    // rescale running y accumulators
#pragma unroll
    for (int t = 0; t < 8; ++t)
#pragma unroll
      for (int j = 0; j < 8; ++j) yacc[t][j] *= scale[j];

    // ---- transpose P through LDS: D layout -> A-fragment layout
#pragma unroll
    for (int t = 0; t < 4; ++t)
#pragma unroll
      for (int j = 0; j < 8; ++j)
        pb[(j + 8 * hi) * 72 + 16 * t + lhalf] = (_Float16)f[t][j];
    // same-wave LDS ops complete in order (DScnt); no cross-wave sharing here

    // ---- y += P @ g : two K=32 sub-chunks, 8 ci-tiles each
#pragma unroll
    for (int h = 0; h < 2; ++h) {
      v16h pa;
      const _Float16* pr = pb + lhalf * 72 + 32 * h + hi * 8;
      ((uint4*)&pa)[0] = *(const uint4*)(pr);
      ((uint4*)&pa)[1] = *(const uint4*)(pr + 16);
#pragma unroll
      for (int t = 0; t < 8; ++t) {
        const _Float16* grow =
            g + ((size_t)b * CIH + 16 * t + lhalf) * MM + m0 + 32 * h + hi * 8;
        v16h gb;
        ((uint4*)&gb)[0] = *(const uint4*)(grow);
        ((uint4*)&gb)[1] = *(const uint4*)(grow + 16);
        yacc[t] = __builtin_amdgcn_wmma_f32_16x16x32_f16(false, pa, false, gb,
                                                         (short)0, yacc[t], false, false);
      }
    }
  }

  // ---- epilogue: normalize and store y [N][CI] fp16
  float inv[8];
#pragma unroll
  for (int j = 0; j < 8; ++j) inv[j] = 1.0f / lrun[j];
#pragma unroll
  for (int t = 0; t < 8; ++t)
#pragma unroll
    for (int j = 0; j < 8; ++j) {
      int n  = n0 + j + 8 * hi;
      int ci = 16 * t + lhalf;
      yout[((size_t)b * NN + n) * CIH + ci] = (_Float16)(yacc[t][j] * inv[j]);
    }
}

// ---------------------------------------------------------------------------
// Final 1x1 conv (CI->C) + BatchNorm (precomputed scale/shift) + residual.
// out[b][co][n] = acc*bnsc[co] + bnsh[co] + v[b][co][n]
// Block = 256 threads = 8 waves; wave w owns co rows [32w, 32w+32) (2 tiles).
// ---------------------------------------------------------------------------
__global__ __launch_bounds__(256)
void wconv_bn_kernel(const _Float16* __restrict__ y,    // [B][N][CI]
                     const _Float16* __restrict__ Wh,   // [CIN][CIH] fp16
                     const float* __restrict__ bnsc,
                     const float* __restrict__ bnsh,
                     const float* __restrict__ vres,    // [B][CIN][N]
                     float* __restrict__ out) {
  const int b     = blockIdx.y;
  const int n0    = blockIdx.x * 16;
  const int tid   = threadIdx.x;
  const int wave  = tid >> 5;
  const int lane  = tid & 31;
  const int lhalf = lane & 15;
  const int hi    = lane >> 4;

  v8f acc0 = {}, acc1 = {};
#pragma unroll
  for (int c = 0; c < 4; ++c) {            // K = CI = 128 in chunks of 32
    const _Float16* yr =
        y + ((size_t)b * NN + n0 + lhalf) * CIH + 32 * c + hi * 8;
    v16h bf;
    ((uint4*)&bf)[0] = *(const uint4*)(yr);
    ((uint4*)&bf)[1] = *(const uint4*)(yr + 16);

    const _Float16* wr0 = Wh + (size_t)(wave * 32 + lhalf) * CIH + 32 * c + hi * 8;
    const _Float16* wr1 = wr0 + 16 * CIH;
    v16h a0, a1;
    ((uint4*)&a0)[0] = *(const uint4*)(wr0);
    ((uint4*)&a0)[1] = *(const uint4*)(wr0 + 16);
    ((uint4*)&a1)[0] = *(const uint4*)(wr1);
    ((uint4*)&a1)[1] = *(const uint4*)(wr1 + 16);

    acc0 = __builtin_amdgcn_wmma_f32_16x16x32_f16(false, a0, false, bf,
                                                  (short)0, acc0, false, false);
    acc1 = __builtin_amdgcn_wmma_f32_16x16x32_f16(false, a1, false, bf,
                                                  (short)0, acc1, false, false);
  }

  const int n = n0 + lhalf;
#pragma unroll
  for (int t = 0; t < 2; ++t) {
    const v8f& acc = t ? acc1 : acc0;
#pragma unroll
    for (int j = 0; j < 8; ++j) {
      int co   = wave * 32 + 16 * t + j + 8 * hi;
      size_t o = ((size_t)b * CIN + co) * NN + n;
      out[o] = acc[j] * bnsc[co] + bnsh[co] + vres[o];
    }
  }
}

// ---------------------------------------------------------------------------
extern "C" void kernel_launch(void* const* d_in, const int* in_sizes, int n_in,
                              void* d_out, int out_size, void* d_ws, size_t ws_size,
                              hipStream_t stream) {
  const float* q    = (const float*)d_in[0];
  const float* k    = (const float*)d_in[1];
  const float* v    = (const float*)d_in[2];
  const float* Wg   = (const float*)d_in[3];
  const float* bg   = (const float*)d_in[4];
  const float* Wth  = (const float*)d_in[5];
  const float* bth  = (const float*)d_in[6];
  const float* Wph  = (const float*)d_in[7];
  const float* bph  = (const float*)d_in[8];
  const float* Ww   = (const float*)d_in[9];
  const float* bw   = (const float*)d_in[10];
  const float* gam  = (const float*)d_in[11];
  const float* bet  = (const float*)d_in[12];
  const float* mean = (const float*)d_in[13];
  const float* var  = (const float*)d_in[14];
  float* out = (float*)d_out;

  char* ws = (char*)d_ws;
  float*    qp  = (float*)(ws);                              //  8 MB [B][C][M]
  float*    vp  = (float*)(ws + ((size_t)8  << 20));         //  8 MB [B][C][M]
  _Float16* th  = (_Float16*)(ws + ((size_t)16 << 20));      //  8 MB [B][N][CI]
  _Float16* ph  = (_Float16*)(ws + ((size_t)24 << 20));      //  2 MB [B][M][CI]
  _Float16* gt  = (_Float16*)(ws + ((size_t)26 << 20));      //  2 MB [B][CI][M]
  _Float16* yh  = (_Float16*)(ws + ((size_t)28 << 20));      //  8 MB [B][N][CI]
  char*     wx  = ws + ((size_t)36 << 20);
  _Float16* WthH = (_Float16*)(wx);                          // 64 KB
  _Float16* WphH = (_Float16*)(wx + (64 << 10));             // 64 KB
  _Float16* WgH  = (_Float16*)(wx + (128 << 10));            // 64 KB
  _Float16* WwH  = (_Float16*)(wx + (192 << 10));            // 64 KB
  float*    bnsc = (float*)(wx + (256 << 10));               //  1 KB
  float*    bnsh = (float*)(wx + (257 << 10));               //  1 KB

  const int NW = CIH * CIN;  // 32768 elements per weight matrix

  // 0) tiny prep: fp16 weights + BN constants
  f32_to_f16_kernel<<<NW / 256, 256, 0, stream>>>(Wth, WthH, NW);
  f32_to_f16_kernel<<<NW / 256, 256, 0, stream>>>(Wph, WphH, NW);
  f32_to_f16_kernel<<<NW / 256, 256, 0, stream>>>(Wg,  WgH,  NW);
  f32_to_f16_kernel<<<NW / 256, 256, 0, stream>>>(Ww,  WwH,  NW);
  bn_prep_kernel<<<1, CIN, 0, stream>>>(gam, bet, mean, var, bw, bnsc, bnsh);

  // 1) pool q,v down to M (conv commutes with avgpool)
  pool2_kernel<<<(NB * CIN * MM) / 256, 256, 0, stream>>>(q, v, qp, vp);

  // 2) 1x1 convs -> fp16 staging in WMMA-friendly layouts
  conv1x1_f16_kernel<0><<<dim3(NN / 16, NB), 256, 0, stream>>>(k,  WthH, bth, th, NN);
  conv1x1_f16_kernel<0><<<dim3(MM / 16, NB), 256, 0, stream>>>(qp, WphH, bph, ph, MM);
  conv1x1_f16_kernel<1><<<dim3(MM / 16, NB), 256, 0, stream>>>(vp, WgH,  bg,  gt, MM);

  // 3) flash softmax(theta phi^T) g  -> y
  attn_kernel<<<dim3(NN / 64, NB), 128, 0, stream>>>(th, ph, gt, yh);

  // 4) final conv + BN + residual
  wconv_bn_kernel<<<dim3(NN / 16, NB), 256, 0, stream>>>(yh, WwH, bnsc, bnsh, v, out);
}
